// TransformerBlock_33011118637687
// MI455X (gfx1250) — matrix operand order repaired and checked by reading, not compiled, not run on hardware
//
#include <hip/hip_runtime.h>
#include <math.h>

// ---------------- types / helpers ----------------
typedef __attribute__((ext_vector_type(16))) __bf16 v16bf;
typedef __attribute__((ext_vector_type(8)))  float  v8f;

struct alignas(16) U4 { unsigned int a, b, c, d; };
union FragU  { U4 q[2]; v16bf v; };
union U4S    { U4 u; unsigned short s[8]; };

#define NTOK   4096
#define DMODEL 1024
#define HEADS  16
#define HD     64
#define FDIM   2048
#define NEXP   8
#define TSEQ   2048

__device__ __forceinline__ float bf2f(unsigned short h) {
    return __uint_as_float(((unsigned int)h) << 16);
}
__device__ __forceinline__ unsigned short f2bf(float f) {
    unsigned int u = __float_as_uint(f);
    u += 0x7FFFu + ((u >> 16) & 1u);   // round-to-nearest-even
    return (unsigned short)(u >> 16);
}

// Async global->LDS DMA (CDNA5, tracked by ASYNCcnt). The low 32 bits of a
// generic pointer into LDS are the wave-relative LDS byte address (ISA 10.2).
__device__ __forceinline__ void async_load_b128(void* lds_dst, const void* gaddr) {
    unsigned lds_off = (unsigned)(uintptr_t)lds_dst;
    asm volatile("global_load_async_to_lds_b128 %0, %1, off"
                 :: "v"(lds_off), "v"(gaddr) : "memory");
}
__device__ __forceinline__ void wait_async0() {
    asm volatile("s_wait_asynccnt 0" ::: "memory");
}

// A-fragment (16x32 bf16, interleaved K layout per ISA 7.12.2):
// lane: row = lane&15, kbase = (lane>>4)*8 ; elems 0-7 = K kbase..kbase+7, elems 8-15 = K kbase+16..kbase+23
__device__ __forceinline__ v16bf load_frag_A(const unsigned short* base, int row, int koff, int stride) {
    int lane = threadIdx.x & 31;
    const unsigned short* p = base + (size_t)(row + (lane & 15)) * stride + koff + (lane >> 4) * 8;
    FragU f;
    f.q[0] = *(const U4*)p;
    f.q[1] = *(const U4*)(p + 16);
    return f.v;
}
// B-fragment (32x16 bf16): lane: col = lane&15, kbase = (lane>>4)*8 ; elems 0-7 = K kbase..kbase+7, elems 8-15 = K kbase+16..kbase+23 (interleaved, matches A)
__device__ __forceinline__ v16bf load_frag_B(const unsigned short* base, int col, int koff, int stride) {
    int lane = threadIdx.x & 31;
    const unsigned short* p = base + (size_t)(col + (lane & 15)) * stride + koff + (lane >> 4) * 8;
    FragU f;
    f.q[0] = *(const U4*)p;
    f.q[1] = *(const U4*)(p + 16);
    return f.v;
}
__device__ __forceinline__ v8f wmma_bf16(v16bf a, v16bf b, v8f c) {
    return __builtin_amdgcn_wmma_f32_16x16x32_bf16(false, a, false, b, (short)0, c, false, false);
}

// ---------------- misc small kernels ----------------
__global__ void k_zero_cnt(int* cnt) {
    if (threadIdx.x < NEXP) cnt[threadIdx.x] = 0;
}

// transpose + cast f32 (K x N) -> bf16 (N x K).  blockDim (32,8), grid (N/32, K/32, batch)
__global__ __launch_bounds__(256) void k_transcast(const float* __restrict__ in,
                                                   unsigned short* __restrict__ out,
                                                   int K, int N) {
    in  += (size_t)blockIdx.z * K * N;
    out += (size_t)blockIdx.z * K * N;
    __shared__ float tile[32][33];
    int tx = threadIdx.x, ty = threadIdx.y;
    int n0 = blockIdx.x * 32, k0 = blockIdx.y * 32;
#pragma unroll
    for (int j = 0; j < 4; j++)
        tile[ty + j * 8][tx] = in[(size_t)(k0 + ty + j * 8) * N + n0 + tx];
    __syncthreads();
#pragma unroll
    for (int j = 0; j < 4; j++)
        out[(size_t)(n0 + ty + j * 8) * K + k0 + tx] = f2bf(tile[tx][ty + j * 8]);
}

// RMSNorm: one block (256 thr) per row of D=1024, output bf16
__global__ __launch_bounds__(256) void k_rmsnorm(const float* __restrict__ x,
                                                 const float* __restrict__ w,
                                                 unsigned short* __restrict__ out) {
    int n = blockIdx.x, t = threadIdx.x;
    __shared__ float red[256];
    float s = 0.f;
#pragma unroll
    for (int i = 0; i < 4; i++) {
        float v = x[(size_t)n * DMODEL + t + i * 256];
        s += v * v;
    }
    red[t] = s;
    __syncthreads();
    for (int k = 128; k > 0; k >>= 1) {
        if (t < k) red[t] += red[t + k];
        __syncthreads();
    }
    float rinv = rsqrtf(red[0] * (1.0f / DMODEL) + 1e-6f);
#pragma unroll
    for (int i = 0; i < 4; i++) {
        int d = t + i * 256;
        out[(size_t)n * DMODEL + d] = f2bf(x[(size_t)n * DMODEL + d] * rinv * w[d]);
    }
}

// ---------------- dense GEMM: C(MxN) = A(MxK,bf16) * BT(NxK,bf16)^T ----------------
// MODE: 0 = f32 out, 1 = bf16 out, 2 = f32 out + f32 residual
// block 256 (8 waves), tile 128x128, K-step 64, double-buffered LDS filled by
// async global->LDS DMA overlapped with WMMA compute.
template <int MODE>
__global__ __launch_bounds__(256) void k_gemm(const unsigned short* __restrict__ A,
                                              const unsigned short* __restrict__ BT,
                                              void* __restrict__ Cv,
                                              const float* __restrict__ resid,
                                              int M, int N, int K) {
    __shared__ unsigned short As[2][128 * 72];
    __shared__ unsigned short Bs[2][128 * 72];
    int tid  = threadIdx.x;
    int wave = tid >> 5, lane = tid & 31;
    int wm = wave >> 1, wn = wave & 1;
    int bm = blockIdx.y * 128, bn = blockIdx.x * 128;

    v8f acc[2][4];
#pragma unroll
    for (int i = 0; i < 2; i++)
#pragma unroll
        for (int j = 0; j < 4; j++) acc[i][j] = (v8f){0, 0, 0, 0, 0, 0, 0, 0};

    // per-thread DMA assignment: 128x64 bf16 tile = 1024 x b128, 4 per thread (each for A and B)
    auto issue_tile = [&](int buf, int k0) {
#pragma unroll
        for (int i = 0; i < 4; i++) {
            int idx = tid + i * 256;
            int row = idx >> 3, kc = (idx & 7) * 8;
            async_load_b128(&As[buf][row * 72 + kc], &A[(size_t)(bm + row) * K + k0 + kc]);
            async_load_b128(&Bs[buf][row * 72 + kc], &BT[(size_t)(bn + row) * K + k0 + kc]);
        }
    };

    int nk = K >> 6;
    issue_tile(0, 0);
    wait_async0();
    __syncthreads();

    for (int kt = 0; kt < nk; kt++) {
        int buf = kt & 1;
        if (kt + 1 < nk) issue_tile(buf ^ 1, (kt + 1) * 64);  // overlap DMA with compute
#pragma unroll
        for (int ks = 0; ks < 2; ks++) {
            v16bf af[2], bf[4];
#pragma unroll
            for (int mf = 0; mf < 2; mf++) af[mf] = load_frag_A(As[buf], wm * 32 + mf * 16, ks * 32, 72);
#pragma unroll
            for (int nf = 0; nf < 4; nf++) bf[nf] = load_frag_B(Bs[buf], wn * 64 + nf * 16, ks * 32, 72);
#pragma unroll
            for (int mf = 0; mf < 2; mf++)
#pragma unroll
                for (int nf = 0; nf < 4; nf++) acc[mf][nf] = wmma_bf16(af[mf], bf[nf], acc[mf][nf]);
        }
        wait_async0();
        __syncthreads();
    }

    int c = lane & 15, half = lane >> 4;
#pragma unroll
    for (int mf = 0; mf < 2; mf++)
#pragma unroll
        for (int nf = 0; nf < 4; nf++)
#pragma unroll
            for (int r = 0; r < 8; r++) {
                int row = bm + wm * 32 + mf * 16 + half * 8 + r;
                int col = bn + wn * 64 + nf * 16 + c;
                float v = acc[mf][nf][r];
                if (MODE == 0) ((float*)Cv)[(size_t)row * N + col] = v;
                if (MODE == 1) ((unsigned short*)Cv)[(size_t)row * N + col] = f2bf(v);
                if (MODE == 2) ((float*)Cv)[(size_t)row * N + col] = v + resid[(size_t)row * N + col];
            }
}

// ---------------- flash attention ----------------
// grid (T/64, B*H), block 128 (4 waves). Each wave owns a 16-q-row strip of the 64-row q tile.
__global__ __launch_bounds__(128) void k_attn(const unsigned short* __restrict__ qkv,
                                              unsigned short* __restrict__ ao) {
    __shared__ unsigned short Qs[64 * 72];
    __shared__ unsigned short Ks[64 * 72];
    __shared__ unsigned short Vt[64 * 72];   // transposed: [d][key]
    __shared__ unsigned short Ps[64 * 72];
    int tid = threadIdx.x, lane = tid & 31, wave = tid >> 5;
    int qt = blockIdx.x, bh = blockIdx.y;
    int b = bh / HEADS, h = bh % HEADS;
    int qbase = b * TSEQ + qt * 64;
    int cc = lane & 15, half = lane >> 4;
    int ws = wave * 16;

#pragma unroll
    for (int i = 0; i < 4; i++) {
        int idx = tid + i * 128;
        int row = idx >> 3, dc = (idx & 7) * 8;
        *(U4*)&Qs[row * 72 + dc] = *(const U4*)&qkv[(size_t)(qbase + row) * 3072 + h * 64 + dc];
    }
    __syncthreads();

    float m_r[8], l_r[8];
    v8f o[4];
#pragma unroll
    for (int r = 0; r < 8; r++) { m_r[r] = -__builtin_inff(); l_r[r] = 0.f; }
#pragma unroll
    for (int nf = 0; nf < 4; nf++) o[nf] = (v8f){0, 0, 0, 0, 0, 0, 0, 0};

    for (int kt = 0; kt <= qt; kt++) {
        __syncthreads();
        int kbase = b * TSEQ + kt * 64;
#pragma unroll
        for (int i = 0; i < 4; i++) {
            int idx = tid + i * 128;
            int row = idx >> 3, dc = (idx & 7) * 8;
            *(U4*)&Ks[row * 72 + dc] = *(const U4*)&qkv[(size_t)(kbase + row) * 3072 + 1024 + h * 64 + dc];
            U4S vv;
            vv.u = *(const U4*)&qkv[(size_t)(kbase + row) * 3072 + 2048 + h * 64 + dc];
#pragma unroll
            for (int j = 0; j < 8; j++) Vt[(dc + j) * 72 + row] = vv.s[j];
        }
        __syncthreads();

        // S = Q K^T for this wave's 16x64 strip
        v8f s[4];
#pragma unroll
        for (int nf = 0; nf < 4; nf++) s[nf] = (v8f){0, 0, 0, 0, 0, 0, 0, 0};
#pragma unroll
        for (int ks = 0; ks < 2; ks++) {
            v16bf aq = load_frag_A(Qs, ws, ks * 32, 72);
#pragma unroll
            for (int nf = 0; nf < 4; nf++)
                s[nf] = wmma_bf16(aq, load_frag_B(Ks, nf * 16, ks * 32, 72), s[nf]);
        }
        bool diag = (kt == qt);
#pragma unroll
        for (int nf = 0; nf < 4; nf++)
#pragma unroll
            for (int r = 0; r < 8; r++) {
                float v = s[nf][r] * 0.125f;
                if (diag) {
                    int rq = qt * 64 + ws + half * 8 + r;
                    int ck = kt * 64 + nf * 16 + cc;
                    if (ck > rq) v = -1e30f;
                }
                s[nf][r] = v;
            }

        // online softmax per row (row spread over 16 lanes of the half-wave)
#pragma unroll
        for (int r = 0; r < 8; r++) {
            float mx = s[0][r];
#pragma unroll
            for (int nf = 1; nf < 4; nf++) mx = fmaxf(mx, s[nf][r]);
#pragma unroll
            for (int m = 1; m < 16; m <<= 1) mx = fmaxf(mx, __shfl_xor(mx, m, 16));
            float mn = fmaxf(m_r[r], mx);
            float alpha = __expf(m_r[r] - mn);
            float psum = 0.f;
#pragma unroll
            for (int nf = 0; nf < 4; nf++) {
                float p = __expf(s[nf][r] - mn);
                psum += p;
                Ps[(ws + half * 8 + r) * 72 + nf * 16 + cc] = f2bf(p);
            }
#pragma unroll
            for (int m = 1; m < 16; m <<= 1) psum += __shfl_xor(psum, m, 16);
            l_r[r] = l_r[r] * alpha + psum;
            m_r[r] = mn;
#pragma unroll
            for (int nf = 0; nf < 4; nf++) o[nf][r] *= alpha;
        }

        // O += P @ V  (B frags from transposed V)
#pragma unroll
        for (int ks = 0; ks < 2; ks++) {
            v16bf ap = load_frag_A(Ps, ws, ks * 32, 72);
#pragma unroll
            for (int nf = 0; nf < 4; nf++)
                o[nf] = wmma_bf16(ap, load_frag_B(Vt, nf * 16, ks * 32, 72), o[nf]);
        }
    }

#pragma unroll
    for (int nf = 0; nf < 4; nf++)
#pragma unroll
        for (int r = 0; r < 8; r++) {
            int row = ws + half * 8 + r;
            int d = nf * 16 + cc;
            ao[(size_t)(qbase + row) * DMODEL + h * 64 + d] = f2bf(o[nf][r] / l_r[r]);
        }
}

// ---------------- router: softmax + top-2 + expert list build ----------------
__global__ __launch_bounds__(64) void k_router(const unsigned short* __restrict__ hn2,
                                               const float* __restrict__ rw,
                                               int* __restrict__ ents,
                                               float* __restrict__ entw,
                                               int* __restrict__ cnt) {
    int n = blockIdx.x, t = threadIdx.x;
    __shared__ float red[64][8];
    float acc[8];
#pragma unroll
    for (int e = 0; e < 8; e++) acc[e] = 0.f;
    for (int d = t; d < DMODEL; d += 64) {
        float x = bf2f(hn2[(size_t)n * DMODEL + d]);
#pragma unroll
        for (int e = 0; e < 8; e++) acc[e] += x * rw[d * 8 + e];
    }
#pragma unroll
    for (int e = 0; e < 8; e++) red[t][e] = acc[e];
    __syncthreads();
    if (t < 8) {
        float s = 0.f;
        for (int i = 0; i < 64; i++) s += red[i][t];
        red[0][t] = s;
    }
    __syncthreads();
    if (t == 0) {
        float lg[8], p[8];
        float mx = red[0][0];
#pragma unroll
        for (int e = 0; e < 8; e++) { lg[e] = red[0][e]; mx = fmaxf(mx, lg[e]); }
#pragma unroll
        for (int e = 0; e < 8; e++) p[e] = __expf(lg[e] - mx);
        int i0 = 0;
#pragma unroll
        for (int e = 1; e < 8; e++) if (p[e] > p[i0]) i0 = e;
        int i1 = (i0 == 0) ? 1 : 0;
#pragma unroll
        for (int e = 0; e < 8; e++) if (e != i0 && p[e] > p[i1]) i1 = e;
        float w0 = p[i0] / (p[i0] + p[i1]);
        float w1 = 1.0f - w0;
        int pos0 = atomicAdd(&cnt[i0], 1);
        ents[i0 * NTOK + pos0] = n * 2 + 0;
        entw[i0 * NTOK + pos0] = w0;
        int pos1 = atomicAdd(&cnt[i1], 1);
        ents[i1 * NTOK + pos1] = n * 2 + 1;
        entw[i1 * NTOK + pos1] = w1;
    }
}

// ---------------- MoE up: up = silu(hn@w1) * (hn@w3), gathered per expert ----------------
// block 128 (4 waves), tile 64 entries x 64 F cols, grid (F/64, NTOK/64, E)
__global__ __launch_bounds__(128) void k_moe_up(const unsigned short* __restrict__ hn2,
                                                const unsigned short* __restrict__ w1T,
                                                const unsigned short* __restrict__ w3T,
                                                const int* __restrict__ ents,
                                                const int* __restrict__ cnt,
                                                unsigned short* __restrict__ upb) {
    int e = blockIdx.z;
    int cnte = cnt[e];
    int m0 = blockIdx.y * 64;
    if (m0 >= cnte) return;
    int n0 = blockIdx.x * 64;
    __shared__ int tokL[64];
    __shared__ unsigned short As[64 * 40], B1[64 * 40], B3[64 * 40];
    int tid = threadIdx.x, lane = tid & 31, wave = tid >> 5;
    int ws = wave * 16;
    if (tid < 64) tokL[tid] = (m0 + tid < cnte) ? ents[e * NTOK + m0 + tid] : -1;
    const unsigned short* w1e = w1T + (size_t)e * FDIM * DMODEL;
    const unsigned short* w3e = w3T + (size_t)e * FDIM * DMODEL;
    v8f a1[4], a3[4];
#pragma unroll
    for (int nf = 0; nf < 4; nf++) { a1[nf] = (v8f){0,0,0,0,0,0,0,0}; a3[nf] = (v8f){0,0,0,0,0,0,0,0}; }

    for (int k0 = 0; k0 < DMODEL; k0 += 32) {
        __syncthreads();
#pragma unroll
        for (int i = 0; i < 2; i++) {
            int idx = tid + i * 128;
            int row = idx >> 2, kc = (idx & 3) * 8;
            int id = tokL[row];
            U4 av{0, 0, 0, 0};
            if (id >= 0) av = *(const U4*)&hn2[(size_t)(id >> 1) * DMODEL + k0 + kc];
            *(U4*)&As[row * 40 + kc] = av;
            *(U4*)&B1[row * 40 + kc] = *(const U4*)&w1e[(size_t)(n0 + row) * DMODEL + k0 + kc];
            *(U4*)&B3[row * 40 + kc] = *(const U4*)&w3e[(size_t)(n0 + row) * DMODEL + k0 + kc];
        }
        __syncthreads();
        v16bf af = load_frag_A(As, ws, 0, 40);
#pragma unroll
        for (int nf = 0; nf < 4; nf++) {
            a1[nf] = wmma_bf16(af, load_frag_B(B1, nf * 16, 0, 40), a1[nf]);
            a3[nf] = wmma_bf16(af, load_frag_B(B3, nf * 16, 0, 40), a3[nf]);
        }
    }
    int c = lane & 15, half = lane >> 4;
#pragma unroll
    for (int nf = 0; nf < 4; nf++)
#pragma unroll
        for (int r = 0; r < 8; r++) {
            int row = ws + half * 8 + r;
            int id = tokL[row];
            if (id >= 0) {
                float h1 = a1[nf][r], h3 = a3[nf][r];
                float up = h1 / (1.f + __expf(-h1)) * h3;
                upb[(size_t)id * FDIM + n0 + nf * 16 + c] = f2bf(up);
            }
        }
}

// ---------------- MoE down: y[(token,slot)] = gate * (up @ w2), gathered ----------------
// grid (D/64, NTOK/64, E)
__global__ __launch_bounds__(128) void k_moe_down(const unsigned short* __restrict__ upb,
                                                  const unsigned short* __restrict__ w2T,
                                                  const int* __restrict__ ents,
                                                  const float* __restrict__ entw,
                                                  const int* __restrict__ cnt,
                                                  float* __restrict__ yslot) {
    int e = blockIdx.z;
    int cnte = cnt[e];
    int m0 = blockIdx.y * 64;
    if (m0 >= cnte) return;
    int n0 = blockIdx.x * 64;
    __shared__ int tokL[64];
    __shared__ float wL[64];
    __shared__ unsigned short As[64 * 40], Bs[64 * 40];
    int tid = threadIdx.x, lane = tid & 31, wave = tid >> 5;
    int ws = wave * 16;
    if (tid < 64) {
        bool valid = (m0 + tid < cnte);
        tokL[tid] = valid ? ents[e * NTOK + m0 + tid] : -1;
        wL[tid] = valid ? entw[e * NTOK + m0 + tid] : 0.f;
    }
    const unsigned short* w2e = w2T + (size_t)e * DMODEL * FDIM;
    v8f acc[4];
#pragma unroll
    for (int nf = 0; nf < 4; nf++) acc[nf] = (v8f){0,0,0,0,0,0,0,0};

    for (int k0 = 0; k0 < FDIM; k0 += 32) {
        __syncthreads();
#pragma unroll
        for (int i = 0; i < 2; i++) {
            int idx = tid + i * 128;
            int row = idx >> 2, kc = (idx & 3) * 8;
            int id = tokL[row];
            U4 av{0, 0, 0, 0};
            if (id >= 0) av = *(const U4*)&upb[(size_t)id * FDIM + k0 + kc];
            *(U4*)&As[row * 40 + kc] = av;
            *(U4*)&Bs[row * 40 + kc] = *(const U4*)&w2e[(size_t)(n0 + row) * FDIM + k0 + kc];
        }
        __syncthreads();
        v16bf af = load_frag_A(As, ws, 0, 40);
#pragma unroll
        for (int nf = 0; nf < 4; nf++)
            acc[nf] = wmma_bf16(af, load_frag_B(Bs, nf * 16, 0, 40), acc[nf]);
    }
    int c = lane & 15, half = lane >> 4;
#pragma unroll
    for (int nf = 0; nf < 4; nf++)
#pragma unroll
        for (int r = 0; r < 8; r++) {
            int row = ws + half * 8 + r;
            int id = tokL[row];
            if (id >= 0) {
                float v = acc[nf][r] * wL[row];
                yslot[(size_t)(id & 1) * NTOK * DMODEL + (size_t)(id >> 1) * DMODEL + n0 + nf * 16 + c] = v;
            }
        }
}

// ---------------- final: out = x_new + y0 + y1 ; aux_loss = 0 (AUX_COEFF == 0) ----------------
__global__ __launch_bounds__(256) void k_final(const float* __restrict__ xn,
                                               const float* __restrict__ y0,
                                               const float* __restrict__ y1,
                                               float* __restrict__ out) {
    int i = blockIdx.x * 256 + threadIdx.x;
    if (i < NTOK * DMODEL) out[i] = xn[i] + y0[i] + y1[i];
    if (i == 0) out[NTOK * DMODEL] = 0.0f;
}

// ---------------- host launch ----------------
extern "C" void kernel_launch(void* const* d_in, const int* in_sizes, int n_in,
                              void* d_out, int out_size, void* d_ws, size_t ws_size,
                              hipStream_t stream) {
    (void)in_sizes; (void)n_in; (void)out_size; (void)ws_size;
    const float* x       = (const float*)d_in[0];
    const float* anorm_w = (const float*)d_in[1];
    const float* fnorm_w = (const float*)d_in[2];
    const float* qkv_w   = (const float*)d_in[3];
    const float* out_w   = (const float*)d_in[4];
    const float* rout_w  = (const float*)d_in[5];
    const float* w1      = (const float*)d_in[6];
    const float* w3      = (const float*)d_in[7];
    const float* w2      = (const float*)d_in[8];
    float* out = (float*)d_out;

    char* ws = (char*)d_ws;
    size_t off = 0;
    auto take = [&](size_t bytes) { size_t r = off; off = (off + bytes + 255) & ~(size_t)255; return r; };
    unsigned short* qkvT = (unsigned short*)(ws + take((size_t)3072 * 1024 * 2));
    unsigned short* outT = (unsigned short*)(ws + take((size_t)1024 * 1024 * 2));
    unsigned short* w1T  = (unsigned short*)(ws + take((size_t)NEXP * FDIM * DMODEL * 2));
    unsigned short* w3T  = (unsigned short*)(ws + take((size_t)NEXP * FDIM * DMODEL * 2));
    unsigned short* w2T  = (unsigned short*)(ws + take((size_t)NEXP * DMODEL * FDIM * 2));
    unsigned short* hn1  = (unsigned short*)(ws + take((size_t)NTOK * DMODEL * 2));
    unsigned short* qkvA = (unsigned short*)(ws + take((size_t)NTOK * 3072 * 2));
    unsigned short* ao   = (unsigned short*)(ws + take((size_t)NTOK * DMODEL * 2));
    float*          xnew = (float*)        (ws + take((size_t)NTOK * DMODEL * 4));
    unsigned short* hn2  = (unsigned short*)(ws + take((size_t)NTOK * DMODEL * 2));
    int*            ents = (int*)          (ws + take((size_t)NEXP * NTOK * 4));
    float*          entw = (float*)        (ws + take((size_t)NEXP * NTOK * 4));
    int*            cnt  = (int*)          (ws + take((size_t)NEXP * 4));
    unsigned short* upb  = (unsigned short*)(ws + take((size_t)NTOK * 2 * FDIM * 2));
    float*          ysl  = (float*)        (ws + take((size_t)2 * NTOK * DMODEL * 4));

    k_zero_cnt<<<1, 32, 0, stream>>>(cnt);

    // weight casts (transposed, bf16 -> fits 192MB L2)
    k_transcast<<<dim3(96, 32, 1), dim3(32, 8), 0, stream>>>(qkv_w, qkvT, 1024, 3072);
    k_transcast<<<dim3(32, 32, 1), dim3(32, 8), 0, stream>>>(out_w, outT, 1024, 1024);
    k_transcast<<<dim3(64, 32, 8), dim3(32, 8), 0, stream>>>(w1, w1T, 1024, 2048);
    k_transcast<<<dim3(64, 32, 8), dim3(32, 8), 0, stream>>>(w3, w3T, 1024, 2048);
    k_transcast<<<dim3(32, 64, 8), dim3(32, 8), 0, stream>>>(w2, w2T, 2048, 1024);

    // attention path
    k_rmsnorm<<<NTOK, 256, 0, stream>>>(x, anorm_w, hn1);
    k_gemm<1><<<dim3(3072 / 128, NTOK / 128), 256, 0, stream>>>(hn1, qkvT, (void*)qkvA, nullptr, NTOK, 3072, 1024);
    k_attn<<<dim3(TSEQ / 64, 2 * HEADS), 128, 0, stream>>>(qkvA, ao);
    k_gemm<2><<<dim3(1024 / 128, NTOK / 128), 256, 0, stream>>>(ao, outT, (void*)xnew, x, NTOK, 1024, 1024);

    // MoE path
    k_rmsnorm<<<NTOK, 256, 0, stream>>>(xnew, fnorm_w, hn2);
    k_router<<<NTOK, 64, 0, stream>>>(hn2, rout_w, ents, entw, cnt);
    k_moe_up<<<dim3(FDIM / 64, NTOK / 64, NEXP), 128, 0, stream>>>(hn2, w1T, w3T, ents, cnt, upb);
    k_moe_down<<<dim3(DMODEL / 64, NTOK / 64, NEXP), 128, 0, stream>>>(upb, w2T, ents, entw, cnt, ysl);

    k_final<<<(NTOK * DMODEL + 255) / 256, 256, 0, stream>>>(xnew, ysl, ysl + (size_t)NTOK * DMODEL, out);
}
